// MyNewGCN_25890062860843
// MI455X (gfx1250) — matrix operand order repaired and checked by reading, not compiled, not run on hardware
//
#include <hip/hip_runtime.h>

#define BM 128
#define BN 128
#define KT 32
#define LDK 40
#define THREADS 256

typedef __attribute__((ext_vector_type(16))) __bf16 bf16x16;
typedef __attribute__((ext_vector_type(4)))  __bf16 bf16x4;
typedef __attribute__((ext_vector_type(8)))  float  f32x8;
typedef __attribute__((ext_vector_type(4)))  float  f32x4;

union FragBF { bf16x16 v; unsigned long long q[4]; };

// C[M,N] = op( A[M,K] @ B[K,N] + bias + res ), fp32 in/out, bf16 WMMA inside.
// Batched over blockIdx.z via strides. All dims assumed multiples of tiles.
// 8 waves: 4 along M x 2 along N; wave tile 32x64 = 2x4 WMMA accumulators.
__global__ __launch_bounds__(THREADS) void gcn_gemm_bf16_wmma(
    const float* __restrict__ A, long strideA, int lda,
    const float* __restrict__ Bm, long strideB, int ldb,
    const float* __restrict__ bias,
    const float* __restrict__ res, long strideR,
    float* __restrict__ C, long strideC,
    int M, int N, int K, int relu)
{
  __shared__ __bf16 sA[BM * LDK];   // [row][k]  (A tile)
  __shared__ __bf16 sB[BN * LDK];   // [col][k]  (B tile, transposed)

  const int tid    = threadIdx.x;
  const int lane   = tid & 31;
  const int half   = lane >> 4;
  const int mr     = lane & 15;
  const int waveId = tid >> 5;
  const int waveM  = waveId & 3;   // 4 waves along M (32 rows each)
  const int waveN  = waveId >> 2;  // 2 waves along N (64 cols each)

  const int blockN = blockIdx.x * BN;
  const int blockM = blockIdx.y * BM;
  const int z      = blockIdx.z;

  A  += (size_t)z * strideA;
  Bm += (size_t)z * strideB;
  C  += (size_t)z * strideC;
  if (res) res += (size_t)z * strideR;

  // ---- per-thread staging geometry (constant across K loop)
  const int c4    = (tid & 7)  << 2;   // A: column within K-slab
  const int rowA0 = tid >> 3;          // A: first of 4 rows (stride 32)
  const int kk0   = tid >> 5;          // B: first of 4 k-rows (stride 8)
  const int n4    = (tid & 31) << 2;   // B: 4 consecutive output columns

  const float* pa = A  + (size_t)(blockM + rowA0) * lda + c4;
  const float* pb = Bm + (size_t)kk0 * ldb + blockN + n4;
  const size_t aStep = (size_t)32 * lda;   // between A chunks
  const size_t bStep = (size_t)8  * ldb;   // between B chunks
  const size_t bAdv  = (size_t)KT * ldb;   // B advance per K-step

  __bf16* stA = &sA[rowA0 * LDK + c4];
  __bf16* stB = &sB[n4 * LDK + kk0];

  f32x8 acc[2][4];
#pragma unroll
  for (int mi = 0; mi < 2; ++mi)
#pragma unroll
    for (int ni = 0; ni < 4; ++ni)
      acc[mi][ni] = (f32x8)(0.0f);

  // ---- software pipeline: prologue load of first K-slab into registers
  f32x4 ra[4];
  f32x4 rb[4];
#pragma unroll
  for (int t = 0; t < 4; ++t) ra[t] = *(const f32x4*)(pa + t * aStep);
#pragma unroll
  for (int t = 0; t < 4; ++t) rb[t] = *(const f32x4*)(pb + t * bStep);

  for (int k0 = 0; k0 < K; k0 += KT) {
    // ---- drain registers -> LDS (packed hw cvt, 8B DS stores for A)
#pragma unroll
    for (int t = 0; t < 4; ++t)
      *(bf16x4*)(stA + t * 32 * LDK) = __builtin_convertvector(ra[t], bf16x4);
#pragma unroll
    for (int t = 0; t < 4; ++t) {
      bf16x4 h = __builtin_convertvector(rb[t], bf16x4);
      __bf16* s = stB + t * 8;
      s[0 * LDK] = h.x;
      s[1 * LDK] = h.y;
      s[2 * LDK] = h.z;
      s[3 * LDK] = h.w;
    }
    __syncthreads();

    // ---- issue next K-slab loads (one clause); wait lands next iteration
    if (k0 + KT < K) {
      pa += KT;
      pb += bAdv;
#pragma unroll
      for (int t = 0; t < 4; ++t) ra[t] = *(const f32x4*)(pa + t * aStep);
#pragma unroll
      for (int t = 0; t < 4; ++t) rb[t] = *(const f32x4*)(pb + t * bStep);
    }

    // ---- A fragments (16x32 bf16: lanes 0-15 hold K0-15, lanes 16-31 hold K16-31)
    FragBF fa[2], fb[4];
    const int kbA = half << 4;               // 0 or 16
#pragma unroll
    for (int mi = 0; mi < 2; ++mi) {
      const __bf16* p = &sA[(waveM * 32 + mi * 16 + mr) * LDK];
      fa[mi].q[0] = *(const unsigned long long*)(p + kbA);
      fa[mi].q[1] = *(const unsigned long long*)(p + kbA + 4);
      fa[mi].q[2] = *(const unsigned long long*)(p + kbA + 8);
      fa[mi].q[3] = *(const unsigned long long*)(p + kbA + 12);
    }
    // ---- B fragments (32x16 bf16: lanes 0-15 hold K0-15, lanes 16-31 hold K16-31)
    const int kbB = half << 4;               // 0 or 16
#pragma unroll
    for (int ni = 0; ni < 4; ++ni) {
      const __bf16* p = &sB[(waveN * 64 + ni * 16 + mr) * LDK + kbB];
      fb[ni].q[0] = *(const unsigned long long*)(p);
      fb[ni].q[1] = *(const unsigned long long*)(p + 4);
      fb[ni].q[2] = *(const unsigned long long*)(p + 8);
      fb[ni].q[3] = *(const unsigned long long*)(p + 12);
    }

#pragma unroll
    for (int mi = 0; mi < 2; ++mi)
#pragma unroll
      for (int ni = 0; ni < 4; ++ni)
        acc[mi][ni] = __builtin_amdgcn_wmma_f32_16x16x32_bf16(
            false, fa[mi].v, false, fb[ni].v, (short)0, acc[mi][ni], false, false);
    __syncthreads();
  }

  // ---- epilogue: C/D layout -> VGPR r holds (M = base + r + 8*half, N = base + mr)
#pragma unroll
  for (int mi = 0; mi < 2; ++mi) {
#pragma unroll
    for (int ni = 0; ni < 4; ++ni) {
      int col  = blockN + waveN * 64 + ni * 16 + mr;
      int rowb = blockM + waveM * 32 + mi * 16 + (half << 3);
      float bv = bias ? bias[col] : 0.0f;
#pragma unroll
      for (int r = 0; r < 8; ++r) {
        int row = rowb + r;
        float v = acc[mi][ni][r] + bv;
        if (res)  v += res[(size_t)row * N + col];
        if (relu) v = fmaxf(v, 0.0f);
        C[(size_t)row * N + col] = v;
      }
    }
  }
  (void)M;
}

// Global max pool over rows: H [B, Nrows, 128] -> out [B, 128]; grid.x = B
__global__ __launch_bounds__(256) void gcn_maxpool(
    const float* __restrict__ H, float* __restrict__ out, int Nrows)
{
  __shared__ float red[256];
  const int b   = blockIdx.x;
  const int col = threadIdx.x & 127;
  const int seg = threadIdx.x >> 7;
  const float* Hb = H + (size_t)b * Nrows * 128;
  float m = -3.402823466e38f;
  for (int r = seg; r < Nrows; r += 2)
    m = fmaxf(m, Hb[(size_t)r * 128 + col]);
  red[threadIdx.x] = m;
  __syncthreads();
  if (seg == 0)
    out[(size_t)b * 128 + col] = fmaxf(red[col], red[128 + col]);
}

// MLP head: one block per output row (3 heads x B). pooled = [6][B][128].
__global__ __launch_bounds__(128) void gcn_head(
    const float* __restrict__ pooled,
    const float* __restrict__ fc2w, const float* __restrict__ fc2b,
    const float* __restrict__ fc3w, const float* __restrict__ fc3b,
    const float* __restrict__ fc4w, const float* __restrict__ fc4b,
    const float* __restrict__ fc5w, const float* __restrict__ fc5b,
    float* __restrict__ out, int Bsz)
{
  __shared__ float d[256];
  __shared__ float h2[128];
  __shared__ float h3[64];
  __shared__ float h4[32];
  const int row  = blockIdx.x;
  const int head = row / Bsz;
  const int b    = row % Bsz;
  const int t    = threadIdx.x;

  d[t]       = pooled[((size_t)head       * Bsz + b) * 128 + t];
  d[128 + t] = pooled[((size_t)(head + 3) * Bsz + b) * 128 + t];
  __syncthreads();

  float s = fc2b[t];
  for (int k = 0; k < 256; ++k) s += d[k] * fc2w[k * 128 + t];
  h2[t] = fmaxf(s, 0.0f);
  __syncthreads();

  if (t < 64) {
    float s3 = fc3b[t];
    for (int k = 0; k < 128; ++k) s3 += h2[k] * fc3w[k * 64 + t];
    h3[t] = fmaxf(s3, 0.0f);
  }
  __syncthreads();

  if (t < 32) {
    float s4 = fc4b[t];
    for (int k = 0; k < 64; ++k) s4 += h3[k] * fc4w[k * 32 + t];
    h4[t] = fmaxf(s4, 0.0f);
  }
  __syncthreads();

  if (t == 0) {
    float s5 = fc5b[0];
    for (int k = 0; k < 32; ++k) s5 += h4[k] * fc5w[k];
    out[row] = s5;
  }
}

static void launch_gemm(const float* A, long sA, int lda,
                        const float* Bm, long sB, int ldb,
                        const float* bias, const float* res, long sR,
                        float* C, long sC, int M, int N, int K, int relu,
                        int batch, hipStream_t stream)
{
  dim3 grid((unsigned)(N / BN), (unsigned)(M / BM), (unsigned)batch);
  gcn_gemm_bf16_wmma<<<grid, THREADS, 0, stream>>>(
      A, sA, lda, Bm, sB, ldb, bias, res, sR, C, sC, M, N, K, relu);
}

extern "C" void kernel_launch(void* const* d_in, const int* in_sizes, int n_in,
                              void* d_out, int out_size, void* d_ws, size_t ws_size,
                              hipStream_t stream) {
  (void)in_sizes; (void)n_in; (void)out_size; (void)ws_size;
  const int Bsz = 4;
  const int NFEAT = 128, NHID = 256, NCLASS = 128;

  const float* solute_ACE      = (const float*)d_in[0];
  const float* solvent_ACE     = (const float*)d_in[1];
  const float* solute_adj      = (const float*)d_in[2];
  const float* solvent_adj_ACE = (const float*)d_in[3];
  const float* solute_NMF      = (const float*)d_in[4];
  const float* solvent_NMF     = (const float*)d_in[5];
  const float* solvent_adj_NMF = (const float*)d_in[6];
  const float* solute_wat      = (const float*)d_in[7];
  const float* solvent_wat     = (const float*)d_in[8];
  const float* solvent_adj_wat = (const float*)d_in[9];
  const float* conv1_w = (const float*)d_in[10];
  const float* conv1_b = (const float*)d_in[11];
  const float* conv2_w = (const float*)d_in[12];
  const float* conv2_b = (const float*)d_in[13];
  const float* fc1_w   = (const float*)d_in[14];
  const float* fc1_b   = (const float*)d_in[15];
  const float* fc2_w   = (const float*)d_in[16];
  const float* fc2_b   = (const float*)d_in[17];
  const float* fc3_w   = (const float*)d_in[18];
  const float* fc3_b   = (const float*)d_in[19];
  const float* fc4_w   = (const float*)d_in[20];
  const float* fc4_b   = (const float*)d_in[21];
  const float* fc5_w   = (const float*)d_in[22];
  const float* fc5_b   = (const float*)d_in[23];

  // workspace layout (floats), sized for the largest branch (N=3072)
  float* ws = (float*)d_ws;
  const size_t NMAX = 3072;
  size_t o = 0;
  float* T1   = ws + o; o += (size_t)Bsz * NMAX * NHID;
  float* H1   = ws + o; o += (size_t)Bsz * NMAX * NHID;
  float* T2   = ws + o; o += (size_t)Bsz * NMAX * NCLASS;
  float* INIT = ws + o; o += (size_t)Bsz * NMAX * NCLASS;
  float* H2   = ws + o; o += (size_t)Bsz * NMAX * NCLASS;
  float* POOL = ws + o; o += (size_t)6 * Bsz * NCLASS;

  struct Br { const float* x; const float* adj; int N; int relu; int slot; };
  const Br brs[6] = {
    { solute_ACE,  solute_adj,      1024, 1, 0 },
    { solute_NMF,  solute_adj,      1024, 0, 1 },   // source omits first relu here
    { solute_wat,  solute_adj,      1024, 1, 2 },
    { solvent_ACE, solvent_adj_ACE, 2048, 1, 3 },
    { solvent_NMF, solvent_adj_NMF, 2560, 1, 4 },
    { solvent_wat, solvent_adj_wat, 3072, 1, 5 },
  };

  for (int i = 0; i < 6; ++i) {
    const Br& br = brs[i];
    const int N = br.N;
    // T1 = X @ conv1_w
    launch_gemm(br.x, (long)N * NFEAT, NFEAT, conv1_w, 0, NHID,
                nullptr, nullptr, 0, T1, (long)N * NHID,
                N, NHID, NFEAT, 0, Bsz, stream);
    // INIT = X @ fc1_w + fc1_b
    launch_gemm(br.x, (long)N * NFEAT, NFEAT, fc1_w, 0, NCLASS,
                fc1_b, nullptr, 0, INIT, (long)N * NCLASS,
                N, NCLASS, NFEAT, 0, Bsz, stream);
    // H1 = act(adj @ T1 + conv1_b)
    launch_gemm(br.adj, (long)N * N, N, T1, (long)N * NHID, NHID,
                conv1_b, nullptr, 0, H1, (long)N * NHID,
                N, NHID, N, br.relu, Bsz, stream);
    // T2 = H1 @ conv2_w
    launch_gemm(H1, (long)N * NHID, NHID, conv2_w, 0, NCLASS,
                nullptr, nullptr, 0, T2, (long)N * NCLASS,
                N, NCLASS, NHID, 0, Bsz, stream);
    // H2 = adj @ T2 + conv2_b + INIT
    launch_gemm(br.adj, (long)N * N, N, T2, (long)N * NCLASS, NCLASS,
                conv2_b, INIT, (long)N * NCLASS, H2, (long)N * NCLASS,
                N, NCLASS, N, 0, Bsz, stream);
    // pooled[slot] = max over rows
    gcn_maxpool<<<Bsz, 256, 0, stream>>>(H2, POOL + (size_t)br.slot * Bsz * NCLASS, N);
  }

  gcn_head<<<3 * Bsz, 128, 0, stream>>>(POOL, fc2_w, fc2_b, fc3_w, fc3_b,
                                        fc4_w, fc4_b, fc5_w, fc5_b,
                                        (float*)d_out, Bsz);
}
